// VqGanCLIPGenerator_83348135346444
// MI455X (gfx1250) — matrix operand order repaired and checked by reading, not compiled, not run on hardware
//
#include <hip/hip_runtime.h>

// ---- problem constants --------------------------------------------------
#define EDIM     256          // embedding dim (K)
#define M_TOKENS 8192         // B*H*W tokens
#define N_CODES  16384        // codebook entries
#define NSTAGE   64           // codebook rows staged in LDS per iteration
#define LDS_PITCH 528         // 256*2 bytes + 16B pad -> conflict-free ds reads
#define NUM_STAGES (N_CODES / NSTAGE)

#if defined(__AMDGCN__) && __has_builtin(__builtin_amdgcn_global_load_async_to_lds_b128) && __has_builtin(__builtin_amdgcn_s_wait_asynccnt)
#define HAVE_ASYNC_LDS 1
#else
#define HAVE_ASYNC_LDS 0
#endif

typedef __attribute__((ext_vector_type(16))) __bf16 v16bf;
typedef __attribute__((ext_vector_type(8)))  float  v8f;
typedef __attribute__((ext_vector_type(4)))  int    v4i;

#if HAVE_ASYNC_LDS
typedef __attribute__((address_space(1))) v4i* gv4i_p;   // global int4*
typedef __attribute__((address_space(3))) v4i* sv4i_p;   // LDS int4*
#endif

struct U32x8 { uint4 lo, hi; };   // 32 bytes, bit-castable to v16bf

__device__ __forceinline__ v16bf make_frag(uint4 lo, uint4 hi) {
    U32x8 t; t.lo = lo; t.hi = hi;
    return __builtin_bit_cast(v16bf, t);
}

__device__ __forceinline__ unsigned short f2bf(float f) {
    unsigned int u = __float_as_uint(f);
    unsigned int r = u + 0x7FFFu + ((u >> 16) & 1u);   // round-to-nearest-even
    return (unsigned short)(r >> 16);
}

// stage 64 codebook rows (32 KB) into padded LDS; async if available
__device__ __forceinline__ void stage_codes(unsigned char* buf,
                                            const unsigned short* src, int tid) {
#pragma unroll
    for (int k = 0; k < 8; ++k) {
        int li  = tid + k * 256;          // 2048 uint4 total
        int row = li >> 5;                // 32 uint4 per code row
        int col = li & 31;
#if HAVE_ASYNC_LDS
        __builtin_amdgcn_global_load_async_to_lds_b128(
            (gv4i_p)(src + (size_t)li * 8),
            (sv4i_p)(buf + row * LDS_PITCH + col * 16),
            0, 0);
#else
        *(uint4*)(buf + row * LDS_PITCH + col * 16) =
            *(const uint4*)(src + (size_t)li * 8);
#endif
    }
}

// ---- kernel 1: z [B,E,H,W] -> X_bf16 [M_TOKENS, EDIM] (token-major) -----
__global__ __launch_bounds__(256)
void vq_prep_x(const float* __restrict__ z, unsigned short* __restrict__ Xbf) {
    int tid = blockIdx.x * 256 + threadIdx.x;          // 0 .. 2M-1
    int t   = tid >> 8;                                // token
    int e   = tid & 255;
    int b   = t >> 10;
    int hw  = t & 1023;
    Xbf[tid] = f2bf(z[b * 262144 + e * 1024 + hw]);
}

// ---- kernel 2: codebook -> bf16 + |c|^2 (one wave per row) --------------
__global__ __launch_bounds__(128)
void vq_prep_c(const float* __restrict__ cb, unsigned short* __restrict__ Cbf,
               float* __restrict__ cnorm) {
    int wave = threadIdx.x >> 5;
    int lane = threadIdx.x & 31;
    int n    = blockIdx.x * 4 + wave;
    const float* row = cb + (size_t)n * EDIM;
    int e0 = lane * 8;

    float s = 0.0f;
    unsigned short v[8];
#pragma unroll
    for (int i = 0; i < 8; ++i) {
        float f = row[e0 + i];
        s += f * f;
        v[i] = f2bf(f);
    }
    uint4 o;
    o.x = (unsigned)v[0] | ((unsigned)v[1] << 16);
    o.y = (unsigned)v[2] | ((unsigned)v[3] << 16);
    o.z = (unsigned)v[4] | ((unsigned)v[5] << 16);
    o.w = (unsigned)v[6] | ((unsigned)v[7] << 16);
    *(uint4*)(Cbf + (size_t)n * EDIM + e0) = o;

#pragma unroll
    for (int off = 16; off >= 1; off >>= 1) s += __shfl_xor(s, off, 32);
    if (lane == 0) cnorm[n] = s;
}

// ---- kernel 3: bf16 WMMA distance sweep + argmin ------------------------
// 8 waves/block, each wave owns 16 tokens, sweeps all 16384 codes.
// Double-buffered LDS staging (async-to-LDS when available).
__global__ __launch_bounds__(256)
void vq_argmin(const unsigned short* __restrict__ Xbf,
               const unsigned short* __restrict__ Cbf,
               const float* __restrict__ cnorm,
               int* __restrict__ aidx) {
    __shared__ __align__(16) unsigned char lds[2 * NSTAGE * LDS_PITCH];

    const int tid   = threadIdx.x;
    const int wave  = tid >> 5;
    const int lane  = tid & 31;
    const int lhalf = lane >> 4;          // 0: rows 0-7 / K-lo, 1: rows 8-15 / K-hi
    const int l16   = lane & 15;
    const int m_base = (blockIdx.x * 8 + wave) * 16;

    // A fragments: 16 tokens x 256 dims = 8 k-tiles of 16x32 bf16 (ISA layout:
    // lane holds row m=l16; lanes 0-15 k=[0..7],[16..23]; lanes 16-31 +8)
    v16bf a[8];
    {
        const unsigned short* xrow = Xbf + (size_t)(m_base + l16) * EDIM;
#pragma unroll
        for (int kt = 0; kt < 8; ++kt) {
            int k0 = kt * 32 + 8 * lhalf;
            uint4 c0 = *(const uint4*)(xrow + k0);
            uint4 c1 = *(const uint4*)(xrow + k0 + 16);
            a[kt] = make_frag(c0, c1);
        }
    }

    float best[8];
    int   bidx[8];
#pragma unroll
    for (int r = 0; r < 8; ++r) { best[r] = 3.4e38f; bidx[r] = 0; }

    stage_codes(lds, Cbf, tid);           // prime buffer 0

    for (int s = 0; s < NUM_STAGES; ++s) {
        unsigned char* buf = lds + (s & 1) * (NSTAGE * LDS_PITCH);
#if HAVE_ASYNC_LDS
        __builtin_amdgcn_s_wait_asynccnt(0);   // own async copies into buf done
#endif
        __syncthreads();                       // everyone's copies done / readers done

        if (s + 1 < NUM_STAGES)                // fill the other buffer while computing
            stage_codes(lds + ((s + 1) & 1) * (NSTAGE * LDS_PITCH),
                        Cbf + (size_t)(s + 1) * NSTAGE * EDIM, tid);
        if (s + 2 < NUM_STAGES)                // warm L2 two stages ahead
            __builtin_prefetch(Cbf + (size_t)(s + 2) * NSTAGE * EDIM + tid * 64, 0, 0);

        const int n0 = s * NSTAGE;
#pragma unroll
        for (int j = 0; j < 4; ++j) {
            // B layout: lane holds column n=l16; lanes 0-15 k=[0..7],[16..23]; lanes 16-31 +8
            const unsigned char* bb = buf + (j * 16 + l16) * LDS_PITCH + 16 * lhalf;
            v16bf bfr[8];
#pragma unroll
            for (int kt = 0; kt < 8; ++kt)     // preload all 8 B fragments
                bfr[kt] = make_frag(*(const uint4*)(bb + kt * 64),
                                    *(const uint4*)(bb + kt * 64 + 32));
            v8f acc = {};
#pragma unroll
            for (int kt = 0; kt < 8; ++kt)     // back-to-back WMMA chain
                acc = __builtin_amdgcn_wmma_f32_16x16x32_bf16(
                        false, a[kt], false, bfr[kt], (short)0, acc, false, false);

            const int   n  = n0 + j * 16 + l16;
            const float cn = cnorm[n];
#pragma unroll
            for (int r = 0; r < 8; ++r) {
                float d = cn - 2.0f * acc[r];  // |x|^2 constant dropped
                if (d < best[r]) { best[r] = d; bidx[r] = n; }
            }
        }
    }

    // reduce across the 16 lanes that share each row set (xor stays in half)
#pragma unroll
    for (int r = 0; r < 8; ++r) {
        float v = best[r];
        int   i = bidx[r];
#pragma unroll
        for (int off = 8; off >= 1; off >>= 1) {
            float ov = __shfl_xor(v, off, 32);
            int   oi = __shfl_xor(i, off, 32);
            if (ov < v || (ov == v && oi < i)) { v = ov; i = oi; }  // first-index tie
        }
        if (l16 == 0) aidx[m_base + lhalf * 8 + r] = i;
    }
}

// ---- kernel 4: gather fp32 codebook rows into [B,E,H,W] -----------------
__global__ __launch_bounds__(256)
void vq_gather(const float* __restrict__ cb, const int* __restrict__ aidx,
               float* __restrict__ out) {
    int tid = blockIdx.x * 256 + threadIdx.x;      // flat out index
    int b   = tid >> 18;
    int e   = (tid >> 10) & 255;
    int hw  = tid & 1023;
    int t   = b * 1024 + hw;
    out[tid] = cb[(size_t)aidx[t] * EDIM + e];
}

// ---- launch -------------------------------------------------------------
extern "C" void kernel_launch(void* const* d_in, const int* in_sizes, int n_in,
                              void* d_out, int out_size, void* d_ws, size_t ws_size,
                              hipStream_t stream) {
    (void)in_sizes; (void)n_in; (void)out_size; (void)ws_size;
    const float* z  = (const float*)d_in[0];
    const float* cb = (const float*)d_in[1];
    float* out = (float*)d_out;

    char* ws = (char*)d_ws;
    unsigned short* Xbf  = (unsigned short*)ws;                                 // 4 MB
    unsigned short* Cbf  = (unsigned short*)(ws + (size_t)M_TOKENS * EDIM * 2); // 8 MB
    float* cnorm = (float*)(ws + (size_t)M_TOKENS * EDIM * 2
                               + (size_t)N_CODES * EDIM * 2);                   // 64 KB
    int*   aidx  = (int*)((char*)cnorm + (size_t)N_CODES * 4);                  // 32 KB

    vq_prep_x <<<(M_TOKENS * EDIM) / 256, 256, 0, stream>>>(z, Xbf);
    vq_prep_c <<<N_CODES / 4, 128, 0, stream>>>(cb, Cbf, cnorm);
    vq_argmin <<<M_TOKENS / (16 * 8), 256, 0, stream>>>(Xbf, Cbf, cnorm, aidx);
    vq_gather <<<(M_TOKENS * EDIM) / 256, 256, 0, stream>>>(cb, aidx, out);
}
